// HeteroGraphSAGE_52183852646755
// MI455X (gfx1250) — matrix operand order repaired and checked by reading, not compiled, not run on hardware
//
#include <hip/hip_runtime.h>
#include <hip/hip_bf16.h>

// ---------------------------------------------------------------------------
// HeteroGraphSAGE + LSTM aggregation on MI455X (gfx1250).
//
// Compute-bound (~328 GFLOP vs ~0.3 GB): all matmuls on v_wmma_f32_16x16x32_bf16.
// Block = 32 nodes (two M=16 WMMA tiles), 8 wave32s split the 640 gate columns
// (5 N-tiles each), K=160 in 5 chunks of 32. Weights are pre-packed into the
// per-lane B-fragment layout and streamed from L2 each step through a 4-deep
// rotating register pipeline (avoids both the round-1 VGPR-spill hoisting and
// the round-2 WMMA WAR-hazard v_nops). Pointer laundering keeps the loads in
// the loop; address_space(1) casts keep them global_load_b128 (not flat).
// Neighbor rows are gathered per time step into a double-buffered LDS tile,
// overlapped with the WMMA work of the current step.
// ---------------------------------------------------------------------------

typedef __attribute__((ext_vector_type(16))) __bf16 bf16x16;
typedef __attribute__((ext_vector_type(8)))  float  v8f;
typedef __attribute__((ext_vector_type(4)))  float  f32x4;

#define N_NODES  50000
#define DEG      16
#define IN_CH    128
#define TYPE_DIM 32
#define F_DIM    160      // = IN_CH + TYPE_DIM = H
#define G_DIM    640      // 4*H
#define OUT_CH   128
#define KT_N     5        // 160/32 K-tiles
#define NT_G     40       // 640/16 gate N-tiles
#define NT_O     8        // 128/16 output N-tiles
#define MB       32       // nodes per block (two 16-row M tiles)

static __device__ __forceinline__ float sigmoid_fast(float x) {
  return 1.0f / (1.0f + __expf(-x));
}

// A fragment (16x32 MxK bf16) from LDS row-major [.][160]:
// lane l: m=l&15, half=l>>4; e<8 -> K=half*8+e, e>=8 -> K=16+half*8+(e-8).
static __device__ __forceinline__ bf16x16 load_a_frag(const __bf16* p) {
  union { bf16x16 v; f32x4 f[2]; } u;
  u.f[0] = *(const f32x4*)(p);
  u.f[1] = *(const f32x4*)(p + 16);
  return u.v;
}
// B fragment: 32 contiguous bytes per lane, forced global address space so the
// laundered integer address still lowers to global_load_b128 (not flat).
static __device__ __forceinline__ bf16x16 load_b_as1(unsigned long long a) {
  typedef const __attribute__((address_space(1))) f32x4* gp4;
  union { bf16x16 v; f32x4 f[2]; } u;
  u.f[0] = ((gp4)a)[0];
  u.f[1] = ((gp4)a)[1];
  return u.v;
}
static __device__ __forceinline__ bf16x16 load_b_frag(const __bf16* p) {
  union { bf16x16 v; f32x4 f[2]; } u;
  u.f[0] = *(const f32x4*)(p);
  u.f[1] = *(const f32x4*)(p + 8);
  return u.v;
}
static __device__ __forceinline__ v8f wmma_bf16(bf16x16 a, bf16x16 b, v8f c) {
  return __builtin_amdgcn_wmma_f32_16x16x32_bf16(false, a, false, b, (short)0,
                                                 c, false, false);
}

// --------------------------- prep: h_in in bf16 ----------------------------
__global__ void hin_kernel(const float* __restrict__ x,
                           const int* __restrict__ types,
                           const float* __restrict__ emb,
                           __bf16* __restrict__ hin) {
  int idx = blockIdx.x * blockDim.x + threadIdx.x;   // over N*160 (exact)
  int node = idx / F_DIM;
  int c = idx - node * F_DIM;
  float v = (c < IN_CH) ? x[node * IN_CH + c]
                        : emb[types[node] * TYPE_DIM + (c - IN_CH)];
  hin[idx] = (__bf16)v;
}

// ----------------- prep: pack W into WMMA B-fragment layout ----------------
// Fragment element [lane][e] = W[nt*16+(lane&15)][kt*32+(e>>3)*16+(lane>>4)*8+(e&7)].
// Packed index = ((nt*KT_N+kt)*32 + lane)*16 + e  (32 contiguous B per lane).
__global__ void pack_w_kernel(const float* __restrict__ W,
                              __bf16* __restrict__ out) {
  int idx = blockIdx.x * blockDim.x + threadIdx.x;
  int e    = idx & 15;
  int lane = (idx >> 4) & 31;
  int blk  = idx >> 9;               // nt*KT_N + kt
  int kt   = blk % KT_N;
  int nt   = blk / KT_N;
  out[idx] = (__bf16)W[(nt * 16 + (lane & 15)) * F_DIM +
                       kt * 32 + (e >> 3) * 16 + (lane >> 4) * 8 + (e & 7)];
}

// ------------------------------- main kernel -------------------------------
__global__ __launch_bounds__(256) void lstm_sage_kernel(
    const __bf16* __restrict__ hin,     // [N][160] bf16
    const int*    __restrict__ src,     // [N*DEG]
    const __bf16* __restrict__ Wih,     // packed gate weights
    const __bf16* __restrict__ Whh,
    const __bf16* __restrict__ Wl,      // packed output weights
    const __bf16* __restrict__ Wr,
    const float*  __restrict__ b_ih,    // [640]
    const float*  __restrict__ b_hh,    // [640]
    const float*  __restrict__ b_l,     // [128]
    float*        __restrict__ out) {   // [N][128]

  __shared__ __align__(16) __bf16 Xs2[2][MB * F_DIM];  // 20,480 B  x_t dbl-buf
  __shared__ __align__(16) float  Gs[MB * G_DIM];      // 81,920 B  gates f32
  __shared__ __align__(16) float  Cs[MB * F_DIM];      // 20,480 B  cell f32
  __shared__ __align__(16) __bf16 Hs[MB * F_DIM];      // 10,240 B  hidden bf16
  __shared__ __align__(16) __bf16 Rs[MB * F_DIM];      // 10,240 B  root feats
  __shared__ float Bs[G_DIM];                          //  2,560 B  gate bias
  __shared__ int   Ss[MB * DEG];                       //  2,048 B  src idx

  const int tid  = threadIdx.x;
  const int lane = tid & 31;
  const int wave = tid >> 5;                 // 0..7
  const int half = lane >> 4;
  const int mrow = lane & 15;
  const int node_base = blockIdx.x * MB;

  // --- block-invariant staging -------------------------------------------
  for (int i = tid; i < MB * DEG; i += 256) {
    int node = node_base + (i >> 4);
    if (node >= N_NODES) node = N_NODES - 1;             // tail clamp
    Ss[i] = src[node * DEG + (i & 15)];
  }
  for (int c = tid; c < MB * (F_DIM / 8); c += 256) {    // root features
    int m = c / (F_DIM / 8), j = c % (F_DIM / 8);
    int node = node_base + m;
    if (node >= N_NODES) node = N_NODES - 1;
    ((f32x4*)(Rs + m * F_DIM))[j] =
        ((const f32x4*)(hin + (long)node * F_DIM))[j];
  }
  for (int i = tid; i < MB * F_DIM; i += 256) {          // h0 = c0 = 0
    Cs[i] = 0.0f;
    Hs[i] = (__bf16)0.0f;
  }
  for (int i = tid; i < G_DIM; i += 256) Bs[i] = b_ih[i] + b_hh[i];
  __syncthreads();

  // --- gather time step 0 into buffer 0 ----------------------------------
  // 32 rows x 20 16B chunks = 640 chunks over 256 threads.
  const int m0 = tid / 20,         o0 = (tid % 20) * 8;
  const int m1 = (tid + 256) / 20, o1 = ((tid + 256) % 20) * 8;
  const int m2 = (tid + 512) / 20, o2 = ((tid + 512) % 20) * 8;  // tid<128 only
  {
    *(f32x4*)(Xs2[0] + m0 * F_DIM + o0) =
        *(const f32x4*)(hin + (long)Ss[m0 * DEG] * F_DIM + o0);
    *(f32x4*)(Xs2[0] + m1 * F_DIM + o1) =
        *(const f32x4*)(hin + (long)Ss[m1 * DEG] * F_DIM + o1);
    if (tid < 128)
      *(f32x4*)(Xs2[0] + m2 * F_DIM + o2) =
          *(const f32x4*)(hin + (long)Ss[m2 * DEG] * F_DIM + o2);
  }
  __syncthreads();

  // Laundered weight base pointers: redefined opaquely every iteration so the
  // compiler streams B fragments from L2 instead of hoisting them into VGPRs.
  unsigned long long wih_p = (unsigned long long)Wih;
  unsigned long long whh_p = (unsigned long long)Whh;
  const unsigned long long lane_ofs = (unsigned long long)(lane * 32);

#pragma unroll 1
  for (int t = 0; t < DEG; ++t) {
    asm volatile("" : "+s"(wih_p), "+s"(whh_p));
    const __bf16* xbuf = Xs2[t & 1];

    // Issue next step's gather loads early; they complete under the WMMAs.
    f32x4 g0, g1, g2;
    const int tp = t + 1;
    if (tp < DEG) {
      g0 = *(const f32x4*)(hin + (long)Ss[m0 * DEG + tp] * F_DIM + o0);
      g1 = *(const f32x4*)(hin + (long)Ss[m1 * DEG + tp] * F_DIM + o1);
      if (tid < 128)
        g2 = *(const f32x4*)(hin + (long)Ss[m2 * DEG + tp] * F_DIM + o2);
    }

    // A fragments for both M tiles (x_t and h_{t-1}), cached for 5 N-tiles.
    bf16x16 ax0[KT_N], ax1[KT_N], ah0[KT_N], ah1[KT_N];
#pragma unroll
    for (int kt = 0; kt < KT_N; ++kt) {
      int ko = kt * 32 + half * 8;
      ax0[kt] = load_a_frag(xbuf + mrow * F_DIM + ko);
      ax1[kt] = load_a_frag(xbuf + (16 + mrow) * F_DIM + ko);
      ah0[kt] = load_a_frag(Hs + mrow * F_DIM + ko);
      ah1[kt] = load_a_frag(Hs + (16 + mrow) * F_DIM + ko);
    }

#pragma unroll 1
    for (int j = 0; j < 5; ++j) {
      const int nt = wave * 5 + j;
      const int col = nt * 16 + mrow;
      const float bias = Bs[col];
      v8f ca, cb;
#pragma unroll
      for (int r = 0; r < 8; ++r) { ca[r] = bias; cb[r] = bias; }

      // Fused K stream: kk 0..4 -> Wih tiles, kk 5..9 -> Whh tiles.
      // B fragments flow through a 4-deep rotating register pipeline, loaded
      // 2 iterations ahead: kills the WMMA WAR-hazard v_nops and gives each
      // load ~2 WMMA-pairs of L2 latency slack.
      const unsigned long long base0 = wih_p + (unsigned long long)(nt * KT_N) * 1024ull + lane_ofs;
      const unsigned long long base1 = whh_p + (unsigned long long)(nt * KT_N) * 1024ull + lane_ofs;
      bf16x16 bb[4];
      bb[0] = load_b_as1(base0);
      bb[1] = load_b_as1(base0 + 1024ull);
#pragma unroll
      for (int kk = 0; kk < 10; ++kk) {
        if (kk < 8) {
          int kn = kk + 2;
          unsigned long long a = (kn < KT_N)
              ? base0 + (unsigned long long)kn * 1024ull
              : base1 + (unsigned long long)(kn - KT_N) * 1024ull;
          bb[(kk + 2) & 3] = load_b_as1(a);
        }
        bf16x16 a0 = (kk < KT_N) ? ax0[kk < KT_N ? kk : 0]
                                 : ah0[kk < KT_N ? 0 : kk - KT_N];
        bf16x16 a1 = (kk < KT_N) ? ax1[kk < KT_N ? kk : 0]
                                 : ah1[kk < KT_N ? 0 : kk - KT_N];
        ca = wmma_bf16(a0, bb[kk & 3], ca);
        cb = wmma_bf16(a1, bb[kk & 3], cb);
      }

#pragma unroll
      for (int r = 0; r < 8; ++r) {
        Gs[(half * 8 + r) * G_DIM + col] = ca[r];
        Gs[(16 + half * 8 + r) * G_DIM + col] = cb[r];
      }
    }

    // Commit the prefetched next-step rows to the other buffer.
    if (tp < DEG) {
      __bf16* nb = Xs2[tp & 1];
      *(f32x4*)(nb + m0 * F_DIM + o0) = g0;
      *(f32x4*)(nb + m1 * F_DIM + o1) = g1;
      if (tid < 128) *(f32x4*)(nb + m2 * F_DIM + o2) = g2;
    }
    __syncthreads();

    // Pointwise LSTM cell update: 32 nodes x 160 ch over 256 threads.
    {
      const int m = tid >> 3;
      const int hb = (tid & 7) * 20;
#pragma unroll
      for (int jj = 0; jj < 20; ++jj) {
        int h = hb + jj;
        float gi = Gs[m * G_DIM + h];
        float gf = Gs[m * G_DIM + 160 + h];
        float gg = Gs[m * G_DIM + 320 + h];
        float go = Gs[m * G_DIM + 480 + h];
        float cc = sigmoid_fast(gf) * Cs[m * F_DIM + h] +
                   sigmoid_fast(gi) * tanhf(gg);
        Cs[m * F_DIM + h] = cc;
        Hs[m * F_DIM + h] = (__bf16)(sigmoid_fast(go) * tanhf(cc));
      }
    }
    __syncthreads();
  }

  // --- epilogue: out = relu(h_last@Wl^T + b_l + h_in@Wr^T) ----------------
  {
    const int col = wave * 16 + mrow;     // nt = wave, 8 waves x 16 cols = 128
    const float bl = b_l[col];
#pragma unroll 1
    for (int mt = 0; mt < 2; ++mt) {
      v8f c;
#pragma unroll
      for (int r = 0; r < 8; ++r) c[r] = bl;
#pragma unroll
      for (int kt = 0; kt < KT_N; ++kt) {
        int ko = kt * 32 + half * 8;
        bf16x16 a = load_a_frag(Hs + (mt * 16 + mrow) * F_DIM + ko);
        bf16x16 b = load_b_frag(Wl + ((wave * KT_N + kt) * 32 + lane) * 16);
        c = wmma_bf16(a, b, c);
      }
#pragma unroll
      for (int kt = 0; kt < KT_N; ++kt) {
        int ko = kt * 32 + half * 8;
        bf16x16 a = load_a_frag(Rs + (mt * 16 + mrow) * F_DIM + ko);
        bf16x16 b = load_b_frag(Wr + ((wave * KT_N + kt) * 32 + lane) * 16);
        c = wmma_bf16(a, b, c);
      }
#pragma unroll
      for (int r = 0; r < 8; ++r) {
        int row = node_base + mt * 16 + half * 8 + r;
        if (row < N_NODES) out[(long)row * OUT_CH + col] = fmaxf(c[r], 0.0f);
      }
    }
  }
}

// ------------------------------ host launcher ------------------------------
extern "C" void kernel_launch(void* const* d_in, const int* in_sizes, int n_in,
                              void* d_out, int out_size, void* d_ws, size_t ws_size,
                              hipStream_t stream) {
  const float* x      = (const float*)d_in[0];
  const int*   types  = (const int*)d_in[1];
  const int*   edge   = (const int*)d_in[2];   // [2][E]; row 0 = src
  const float* emb    = (const float*)d_in[3];
  const float* W_ih   = (const float*)d_in[4];
  const float* W_hh   = (const float*)d_in[5];
  const float* b_ih   = (const float*)d_in[6];
  const float* b_hh   = (const float*)d_in[7];
  const float* W_l    = (const float*)d_in[8];
  const float* b_l    = (const float*)d_in[9];
  const float* W_r    = (const float*)d_in[10];
  float* out = (float*)d_out;
  (void)in_sizes; (void)n_in; (void)out_size; (void)ws_size;

  // Workspace layout (bytes, 16-aligned).
  char* ws = (char*)d_ws;
  __bf16* hin  = (__bf16*)(ws + 0);                 // 16,000,000 B
  __bf16* wihp = (__bf16*)(ws + 16000000);          //    204,800 B
  __bf16* whhp = (__bf16*)(ws + 16204800);          //    204,800 B
  __bf16* wlp  = (__bf16*)(ws + 16409600);          //     40,960 B
  __bf16* wrp  = (__bf16*)(ws + 16450560);          //     40,960 B

  hin_kernel<<<(N_NODES * F_DIM) / 256, 256, 0, stream>>>(x, types, emb, hin);
  pack_w_kernel<<<(NT_G * KT_N * 512) / 256, 256, 0, stream>>>(W_ih, wihp);
  pack_w_kernel<<<(NT_G * KT_N * 512) / 256, 256, 0, stream>>>(W_hh, whhp);
  pack_w_kernel<<<(NT_O * KT_N * 512) / 256, 256, 0, stream>>>(W_l, wlp);
  pack_w_kernel<<<(NT_O * KT_N * 512) / 256, 256, 0, stream>>>(W_r, wrp);

  lstm_sage_kernel<<<(N_NODES + MB - 1) / MB, 256, 0, stream>>>(
      hin, edge, wihp, whhp, wlp, wrp, b_ih, b_hh, b_l, out);
}
